// ChEst_23845658428009
// MI455X (gfx1250) — matrix operand order repaired and checked by reading, not compiled, not run on hardware
//
#include <hip/hip_runtime.h>
#include <hip/hip_bf16.h>
#include <math.h>

#define HH 8192      // hidden size
#define NT 64        // time steps
#define KCH 16       // K-chunks per recurrent matvec (occupancy)
#define KLEN (HH / KCH)   // 512

typedef __attribute__((ext_vector_type(16))) __bf16 v16bf;
typedef __attribute__((ext_vector_type(8)))  __bf16 v8bf;
typedef __attribute__((ext_vector_type(8)))  float  v8f;

// ---- helpers ---------------------------------------------------------------

// f32 -> bf16 bits, round-to-nearest-even
static __device__ __forceinline__ unsigned f2bf(float f) {
    unsigned u = __float_as_uint(f);
    return (u + 0x7FFFu + ((u >> 16) & 1u)) >> 16;
}
static __device__ __forceinline__ void pack8(v16bf& frag, int base, float4 a, float4 b) {
    frag[base + 0] = (__bf16)a.x; frag[base + 1] = (__bf16)a.y;
    frag[base + 2] = (__bf16)a.z; frag[base + 3] = (__bf16)a.w;
    frag[base + 4] = (__bf16)b.x; frag[base + 5] = (__bf16)b.y;
    frag[base + 6] = (__bf16)b.z; frag[base + 7] = (__bf16)b.w;
}

// ---- phase 0a: W_hh fp32 -> bf16 (makes the recurrent weight L2-resident) --

__global__ void cvt_whh_bf16(const float* __restrict__ src,
                             unsigned short* __restrict__ dst) {
    size_t i = ((size_t)blockIdx.x * blockDim.x + threadIdx.x) * 8;   // 8 elems/thread
    float4 a = *(const float4*)(src + i);
    float4 b = *(const float4*)(src + i + 4);
    uint4 o;
    o.x = f2bf(a.x) | (f2bf(a.y) << 16);
    o.y = f2bf(a.z) | (f2bf(a.w) << 16);
    o.z = f2bf(b.x) | (f2bf(b.y) << 16);
    o.w = f2bf(b.z) | (f2bf(b.w) << 16);
    *(uint4*)(dst + i) = o;
}

// ---- phase 0b: zero-init hidden state buffers ------------------------------

__global__ void init_h_bf(unsigned short* __restrict__ h) {
    h[blockIdx.x * blockDim.x + threadIdx.x] = 0;      // bf16 +0.0
}
__global__ void init_h_f32(float* __restrict__ h) {
    h[blockIdx.x * blockDim.x + threadIdx.x] = 0.0f;
}

// ---- phase 1: P = x @ W_ih^T  (M=64, N=K=8192) via bf16 WMMA ---------------
// One wave per 16-column N-tile; that wave computes all 4 M-tiles (M=64),
// reusing the B fragment (W_ih row data) across 4 WMMAs per K step.
// W_ih is read exactly once -> stream fp32 from HBM, convert in-register.

__global__ void __launch_bounds__(256)
phase1_gemm(const float* __restrict__ x,      // (64, 8192) fp32
            const float* __restrict__ Wih,    // (8192, 8192) fp32, row-major
            float* __restrict__ P) {          // (64, 8192) fp32 out
    const int lane  = threadIdx.x & 31;
    const int wave  = blockIdx.x * (blockDim.x >> 5) + (threadIdx.x >> 5); // 0..511
    const int nbase = wave * 16;

    const int col   = lane & 15;   // B column / C-D column
    const int khalf = lane >> 4;   // 0 or 1
    const int arow  = lane & 15;   // A row within tile
    const int akb   = khalf * 8;   // A K sub-run base

    v8f acc0 = {}, acc1 = {}, acc2 = {}, acc3 = {};

    // B[k][n] = W_ih[n][k]  -> column `nbase+col` of B is row `nbase+col` of W_ih
    const float* wrow = Wih + (size_t)(nbase + col) * HH;

    for (int k0 = 0; k0 < HH; k0 += 32) {
        // B fragment: W_ih row, K runs khalf*8..+7 and 16+khalf*8..+7
        const float4* wp4 = (const float4*)(wrow + k0 + khalf * 8);
        v16bf bfrag;
        pack8(bfrag, 0, wp4[0], wp4[1]);
        pack8(bfrag, 8, wp4[4], wp4[5]);

        // 4 M-tiles share this B fragment
        {
            const float4* xp = (const float4*)(x + (size_t)(0 * 16 + arow) * HH + k0);
            v16bf af;
            pack8(af, 0, xp[akb / 4],     xp[akb / 4 + 1]);
            pack8(af, 8, xp[akb / 4 + 4], xp[akb / 4 + 5]);
            acc0 = __builtin_amdgcn_wmma_f32_16x16x32_bf16(false, af, false, bfrag,
                                                           (short)0, acc0, false, false);
        }
        {
            const float4* xp = (const float4*)(x + (size_t)(1 * 16 + arow) * HH + k0);
            v16bf af;
            pack8(af, 0, xp[akb / 4],     xp[akb / 4 + 1]);
            pack8(af, 8, xp[akb / 4 + 4], xp[akb / 4 + 5]);
            acc1 = __builtin_amdgcn_wmma_f32_16x16x32_bf16(false, af, false, bfrag,
                                                           (short)0, acc1, false, false);
        }
        {
            const float4* xp = (const float4*)(x + (size_t)(2 * 16 + arow) * HH + k0);
            v16bf af;
            pack8(af, 0, xp[akb / 4],     xp[akb / 4 + 1]);
            pack8(af, 8, xp[akb / 4 + 4], xp[akb / 4 + 5]);
            acc2 = __builtin_amdgcn_wmma_f32_16x16x32_bf16(false, af, false, bfrag,
                                                           (short)0, acc2, false, false);
        }
        {
            const float4* xp = (const float4*)(x + (size_t)(3 * 16 + arow) * HH + k0);
            v16bf af;
            pack8(af, 0, xp[akb / 4],     xp[akb / 4 + 1]);
            pack8(af, 8, xp[akb / 4 + 4], xp[akb / 4 + 5]);
            acc3 = __builtin_amdgcn_wmma_f32_16x16x32_bf16(false, af, false, bfrag,
                                                           (short)0, acc3, false, false);
        }
    }

    // D layout: VGPR v -> row v + 8*khalf, col = lane&15
    #pragma unroll
    for (int v = 0; v < 8; ++v) {
        int m = v + 8 * khalf;
        P[(size_t)(0 * 16 + m) * HH + nbase + col] = acc0[v];
        P[(size_t)(1 * 16 + m) * HH + nbase + col] = acc1[v];
        P[(size_t)(2 * 16 + m) * HH + nbase + col] = acc2[v];
        P[(size_t)(3 * 16 + m) * HH + nbase + col] = acc3[v];
    }
}

// ---- phase 2a: partial matvec  partial[c][r] = sum_{k in chunk c} W[r][k]*h[k]
// WMMA matvec: broadcast h (already bf16) into all 16 B columns;
// A = 16x32 W_hh tile (bf16). One wave per (16-row tile, K-chunk):
// 8192 waves/step, 1 KB of W per WMMA; hot loop = 3 loads + 1 WMMA.

__global__ void __launch_bounds__(256)
rnn_step_wmma(const unsigned short* __restrict__ Whh,   // (8192, 8192) bf16
              const unsigned short* __restrict__ h_bf,  // (8192,) bf16
              float* __restrict__ partial) {            // (KCH, 8192) fp32
    const int lane  = threadIdx.x & 31;
    const int w     = blockIdx.x * (blockDim.x >> 5) + (threadIdx.x >> 5); // 0..8191
    const int rt    = w & (HH / 16 - 1);     // row tile 0..511
    const int chunk = w >> 9;                // 0..15
    const int rbase = rt * 16;
    const int kbase = chunk * KLEN;

    const int arow = lane & 15;
    const int half = lane >> 4;
    const int akb  = half * 16;              // A K run base
    const int bkb  = half * 16;              // B K run base

    const unsigned short* wrow = Whh + (size_t)(rbase + arow) * HH;
    const __bf16* hb = (const __bf16*)h_bf;

    v8f acc0 = {}, acc1 = {};
    #pragma unroll 2
    for (int kk = 0; kk < KLEN; kk += 64) {  // 8 iters, 2 independent WMMA chains
        #pragma unroll
        for (int u = 0; u < 2; ++u) {
            const int k0 = kbase + kk + u * 32;
            // A: W_hh[rbase+arow][k0 + akb..akb+15]
            v8bf a0 = *(const v8bf*)(wrow + k0 + akb);
            v8bf a1 = *(const v8bf*)(wrow + k0 + akb + 8);
            v16bf afrag = __builtin_shufflevector(a0, a1,
                0, 1, 2, 3, 4, 5, 6, 7, 8, 9, 10, 11, 12, 13, 14, 15);
            // B: h broadcast into every column; lane holds K run bkb..bkb+15
            v16bf bfrag = *(const v16bf*)(hb + k0 + bkb);
            if (u == 0)
                acc0 = __builtin_amdgcn_wmma_f32_16x16x32_bf16(false, afrag, false, bfrag,
                                                               (short)0, acc0, false, false);
            else
                acc1 = __builtin_amdgcn_wmma_f32_16x16x32_bf16(false, afrag, false, bfrag,
                                                               (short)0, acc1, false, false);
        }
    }

    // All 16 D columns are identical; column 0 lives in lanes 0 (rows 0-7)
    // and 16 (rows 8-15).
    if ((lane & 15) == 0) {
        float* p = partial + (size_t)chunk * HH + rbase + half * 8;
        #pragma unroll
        for (int v = 0; v < 8; ++v) p[v] = acc0[v] + acc1[v];
    }
}

// ---- phase 2b: reduce partials, add P+b, tanh, write h_next (bf16) + output

__global__ void __launch_bounds__(256)
rnn_reduce(const float* __restrict__ partial,
           const float* __restrict__ Pt,
           const float* __restrict__ bias,
           unsigned short* __restrict__ h_next_bf,
           float* __restrict__ out_t) {
    const int r = blockIdx.x * blockDim.x + threadIdx.x;
    float s = Pt[r] + bias[r];
    #pragma unroll
    for (int c = 0; c < KCH; ++c) s += partial[(size_t)c * HH + r];
    float v = tanhf(s);
    out_t[r] = v;
    h_next_bf[r] = (unsigned short)f2bf(v);
}

// fallback if workspace can't hold the bf16 weight copy: stream fp32 W_hh
__global__ void __launch_bounds__(256)
rnn_step_f32(const float* __restrict__ Whh,
             const float* __restrict__ Pt,
             const float* __restrict__ bias,
             const float* __restrict__ h_prev,
             float* __restrict__ h_next,
             float* __restrict__ out_t) {
    const int lane = threadIdx.x & 31;
    const int row  = blockIdx.x * (blockDim.x >> 5) + (threadIdx.x >> 5);
    const float* wrow = Whh + (size_t)row * HH;

    float sum = 0.0f;
    #pragma unroll 4
    for (int c = 0; c < HH / 128; ++c) {              // 64 iterations
        const int kb = (c * 32 + lane) * 4;
        float4 w = *(const float4*)(wrow + kb);
        float4 h = *(const float4*)(h_prev + kb);
        sum += w.x * h.x + w.y * h.y + w.z * h.z + w.w * h.w;
    }
    #pragma unroll
    for (int off = 16; off > 0; off >>= 1)
        sum += __shfl_xor(sum, off, 32);
    if (lane == 0) {
        float v = tanhf(Pt[row] + bias[row] + sum);
        h_next[row] = v;
        out_t[row]  = v;
    }
}

// ---- driver ----------------------------------------------------------------

extern "C" void kernel_launch(void* const* d_in, const int* in_sizes, int n_in,
                              void* d_out, int out_size, void* d_ws, size_t ws_size,
                              hipStream_t stream) {
    const float* x   = (const float*)d_in[0];   // (64, 8192)
    const float* Wih = (const float*)d_in[1];   // (8192, 8192)
    const float* Whh = (const float*)d_in[2];   // (8192, 8192)
    const float* b   = (const float*)d_in[3];   // (8192,)
    float* out = (float*)d_out;                 // (4096, 128) == (64, 8192) flat

    const size_t whh_bf_bytes = (size_t)HH * HH * sizeof(unsigned short); // 128 MB
    const size_t p_bytes      = (size_t)NT * HH * sizeof(float);          // 2 MB
    const size_t hf_bytes     = (size_t)HH * sizeof(float);
    const size_t hb_bytes     = (size_t)HH * sizeof(unsigned short);
    const size_t part_bytes   = (size_t)KCH * HH * sizeof(float);         // 512 KB

    char* ws = (char*)d_ws;
    const bool use_bf16 =
        ws_size >= whh_bf_bytes + p_bytes + 2 * hf_bytes + 2 * hb_bytes + part_bytes;

    unsigned short* whh_bf = (unsigned short*)ws;
    float* P  = use_bf16 ? (float*)(ws + whh_bf_bytes) : (float*)ws;
    float* hAf = (float*)((char*)P + p_bytes);       // fp32 h (fallback path)
    float* hBf = hAf + HH;
    float* partial = hBf + HH;
    unsigned short* hAb = (unsigned short*)(partial + (size_t)KCH * HH);  // bf16 h
    unsigned short* hBb = hAb + HH;

    if (use_bf16) {
        // 64M elems, 8 per thread -> 8M threads
        cvt_whh_bf16<<<((size_t)HH * HH / 8) / 256, 256, 0, stream>>>(Whh, whh_bf);
        init_h_bf<<<HH / 256, 256, 0, stream>>>(hAb);
    } else {
        init_h_f32<<<HH / 256, 256, 0, stream>>>(hAf);
    }

    // Phase 1: 512 N-tile waves, 8 waves/block -> 64 blocks
    phase1_gemm<<<(HH / 16) / 8, 256, 0, stream>>>(x, Wih, P);

    // Phase 2: 64 dependent steps; kernel boundaries provide the cross-step sync
    for (int t = 0; t < NT; ++t) {
        if (use_bf16) {
            const unsigned short* hp = (t & 1) ? hBb : hAb;
            unsigned short*       hn = (t & 1) ? hAb : hBb;
            rnn_step_wmma<<<HH / 8, 256, 0, stream>>>(whh_bf, hp, partial);
            rnn_reduce<<<HH / 256, 256, 0, stream>>>(partial, P + (size_t)t * HH, b,
                                                     hn, out + (size_t)t * HH);
        } else {
            const float* hp = (t & 1) ? hBf : hAf;
            float*       hn = (t & 1) ? hAf : hBf;
            rnn_step_f32<<<HH / 8, 256, 0, stream>>>(Whh, P + (size_t)t * HH, b,
                                                     hp, hn, out + (size_t)t * HH);
        }
    }
}
